// Dota2Eq3Embed_75977971466566
// MI455X (gfx1250) — matrix operand order repaired and checked by reading, not compiled, not run on hardware
//
#include <hip/hip_runtime.h>

typedef __attribute__((ext_vector_type(16))) _Float16 v16h;
typedef __attribute__((ext_vector_type(8)))  _Float16 v8h;
typedef __attribute__((ext_vector_type(8)))  float    v8f;

#define NEMBED     128
#define EMBED_DIM  64
#define HID_DIM    128
#define OUT_DIM    128
#define NCLS       2
#define BATCH      2048
#define TEAM       5
#define NOPS       8
#define KTOT       512      // EMBED_DIM * NOPS
#define NPOS       125      // 5^3 valid rows; padded to 128
#define MTILES     8        // 128 rows / 16
#define NTILES     8        // 128 cols / 16
#define KTILES     16       // 512 / 32
#define CHUNKS     4
#define KT_PER_CHUNK 4      // KTILES / CHUNKS
#define D_PER_CHUNK  16     // 128 K per chunk / 8 ops
#define USTRIDE    136      // 128 + 8 halfs pad (keeps 16B row alignment: 136*2=272)

// Packed B size per net: KTILES * NTILES * 32 lanes * 16 halfs
#define BPACK_PER_NET (KTILES * NTILES * 32 * 16)

// ---------------------------------------------------------------------------
// Kernel 1: pack coefs (f32, layout (d=64, s=128, b=8)) into f16 WMMA-B
// fragment order: [net][kt][nt][lane][elem], elem e of lane l holds
// B[k = kt*32 + e + 16*(l>>4)][s = nt*16 + (l&15)], with k -> (d = k>>3, b = k&7).
// ---------------------------------------------------------------------------
__global__ __launch_bounds__(256) void pack_coefs_kernel(
    const float* __restrict__ coefs1, const float* __restrict__ coefs2,
    _Float16* __restrict__ bpack) {
  int id = blockIdx.x * blockDim.x + threadIdx.x;   // 0 .. 2*BPACK_PER_NET-1
  int e    = id & 15;
  int lane = (id >> 4) & 31;
  int nt   = (id >> 9) & 7;
  int kt   = (id >> 12) & 15;
  int net  = id >> 16;
  int kl = e + ((lane >> 4) << 4);                  // K within 32-wide tile
  int s  = nt * 16 + (lane & 15);
  int kk = kt * 32 + kl;                            // 0..511
  int d  = kk >> 3, b = kk & 7;
  const float* c = net ? coefs2 : coefs1;
  bpack[id] = (_Float16)c[(d * HID_DIM + s) * NOPS + b];
}

// ---------------------------------------------------------------------------
// Kernel 2: per-(n,net) fused eq3 layer:
//   U(128x512,f16) = feature rows per (i,j,k);  out = U @ Cpack (WMMA f32 acc)
//   H[n,net,s] = mean_{p<125} relu(out[p,s] + bias[s])
// ---------------------------------------------------------------------------
__global__ __launch_bounds__(256) void eq3_main_kernel(
    const int* __restrict__ x, const float* __restrict__ embed,
    const _Float16* __restrict__ bpack,
    const float* __restrict__ bias1, const float* __restrict__ bias2,
    float* __restrict__ Hout) {
  __shared__ float a_s[TEAM][EMBED_DIM];
  __shared__ float S1s[EMBED_DIM], S2s[EMBED_DIM], S3s[EMBED_DIM];
  __shared__ _Float16 U[128][USTRIDE];              // one 128x128 K-chunk of U

  const int n = blockIdx.x, net = blockIdx.y;
  const int tid  = threadIdx.x;
  const int wave = tid >> 5;                        // 8 waves, wave == N-tile
  const int lane = tid & 31;
  const int hi   = lane >> 4;                       // lane half (wave32)
  const int lo   = lane & 15;

  // ---- Stage A: gather + ReLU embedding rows, then S, S^2, S^3 per channel
  for (int t = tid; t < TEAM * EMBED_DIM; t += 256) {
    int i = t >> 6, d = t & 63;
    int idx = x[(n * 2 + net) * TEAM + i];
    float v = embed[idx * EMBED_DIM + d];
    a_s[i][d] = v > 0.f ? v : 0.f;
  }
  __syncthreads();
  if (tid < EMBED_DIM) {
    float S = a_s[0][tid] + a_s[1][tid] + a_s[2][tid] + a_s[3][tid] + a_s[4][tid];
    S1s[tid] = S; S2s[tid] = S * S; S3s[tid] = S * S * S;
  }

  v8f acc[MTILES] = {};                             // f32 accumulators, 8 M-tiles
  const _Float16* bbase = bpack + (size_t)net * BPACK_PER_NET;

  for (int c = 0; c < CHUNKS; ++c) {
    // Prefetch next chunk's B fragments (emits global_prefetch_b8)
    if (c + 1 < CHUNKS) {
      const _Float16* pf =
          bbase + (size_t)(c + 1) * KT_PER_CHUNK * NTILES * 32 * 16 + (size_t)tid * 64;
      __builtin_prefetch(pf, 0, 1);
    }
    __syncthreads();                                // U safe to overwrite

    // ---- Stage B: build U chunk (cols kk in [c*128, c*128+128)), 8 ops/entry
    for (int it = 0; it < 8; ++it) {
      int pair = tid + it * 256;                    // 2048 (p, d_local) pairs
      int p  = pair >> 4;                           // 0..127 (>=125 zero-pad)
      int dl = pair & 15;
      int d  = c * D_PER_CHUNK + dl;
      int i = p / 25, j = (p / 5) % 5, k = p % 5;
      float ai = 0.f, aj = 0.f, ak = 0.f, S = 0.f, S2 = 0.f, S3 = 0.f;
      if (p < NPOS) {
        ai = a_s[i][d]; aj = a_s[j][d]; ak = a_s[k][d];
        S = S1s[d]; S2 = S2s[d]; S3 = S3s[d];
      }
      float jk = aj * ak;
      v8h vals;
      vals[0] = (_Float16)(ai * jk);        // t
      vals[1] = (_Float16)(S * jk);         // sum_i
      vals[2] = (_Float16)(S * ai * ak);    // sum_j
      vals[3] = (_Float16)(S * ai * aj);    // sum_k
      vals[4] = (_Float16)(S2 * ak);        // sum_ij
      vals[5] = (_Float16)(S2 * aj);        // sum_ik
      vals[6] = (_Float16)(S2 * ai);        // sum_jk
      vals[7] = (_Float16)(S3);             // sum_ijk
      *(v8h*)(&U[p][dl * 8]) = vals;        // 16B-aligned LDS store
    }
    __syncthreads();

    // ---- Stage C: WMMA over this chunk's 4 k-tiles; wave owns N-tile `wave`
    for (int ktl = 0; ktl < KT_PER_CHUNK; ++ktl) {
      int kt = c * KT_PER_CHUNK + ktl;
      const _Float16* bp =
          bbase + (((size_t)kt * NTILES + wave) * 32 + lane) * 16;
      v16h bfrag = *(const v16h*)bp;                // 32B contiguous fragment
      #pragma unroll
      for (int mt = 0; mt < MTILES; ++mt) {
        // A layout (16x32) matching packed B: lane lo = M row; VGPR v -> K=2v+16*hi;
        // element pairs are K-contiguous -> b32 reads.
        union { v16h v; unsigned u[8]; } af;
        const _Float16* urow = &U[mt * 16 + lo][ktl * 32];
        #pragma unroll
        for (int v2 = 0; v2 < 8; ++v2) {
          int kloc = 2 * v2 + 16 * hi;
          af.u[v2] = *(const unsigned*)(urow + kloc);
        }
        acc[mt] = __builtin_amdgcn_wmma_f32_16x16x32_f16(
            false, af.v, false, bfrag, (short)0, acc[mt], false, false);
      }
    }
  }

  // ---- Stage D: bias + ReLU + masked mean over the 125 valid positions
  const float* bias = net ? bias2 : bias1;
  int s = wave * 16 + lo;
  float bv = bias[s];
  float sum = 0.f;
  #pragma unroll
  for (int mt = 0; mt < MTILES; ++mt) {
    #pragma unroll
    for (int r = 0; r < 8; ++r) {
      int m = mt * 16 + r + 8 * hi;                 // C/D layout: row = r + 8*hi
      float v = acc[mt][r] + bv;
      v = v > 0.f ? v : 0.f;
      if (m < NPOS) sum += v;
    }
  }
  sum += __shfl_xor(sum, 16, 32);                   // combine the two lane halves
  if (hi == 0)
    Hout[((size_t)n * 2 + net) * HID_DIM + s] = sum * (1.f / 125.f);
}

// ---------------------------------------------------------------------------
// Kernel 3: epilogue: t_e = relu(H @ wout + bout); out = [t1e|t2e] @ fcw + fcb
// ---------------------------------------------------------------------------
__global__ __launch_bounds__(128) void head_kernel(
    const float* __restrict__ H,
    const float* __restrict__ wout1, const float* __restrict__ bout1,
    const float* __restrict__ wout2, const float* __restrict__ bout2,
    const float* __restrict__ fcw, const float* __restrict__ fcb,
    float* __restrict__ out) {
  __shared__ float h1[HID_DIM], h2[HID_DIM];
  __shared__ float red0[128], red1[128];
  int n = blockIdx.x, o = threadIdx.x;
  h1[o] = H[((size_t)n * 2 + 0) * HID_DIM + o];
  h2[o] = H[((size_t)n * 2 + 1) * HID_DIM + o];
  __syncthreads();
  float a1 = bout1[o], a2 = bout2[o];
  for (int s = 0; s < HID_DIM; ++s) {
    a1 += h1[s] * wout1[s * OUT_DIM + o];
    a2 += h2[s] * wout2[s * OUT_DIM + o];
  }
  float t1 = a1 > 0.f ? a1 : 0.f;
  float t2 = a2 > 0.f ? a2 : 0.f;
  red0[o] = t1 * fcw[o * NCLS + 0] + t2 * fcw[(OUT_DIM + o) * NCLS + 0];
  red1[o] = t1 * fcw[o * NCLS + 1] + t2 * fcw[(OUT_DIM + o) * NCLS + 1];
  __syncthreads();
  for (int st = 64; st > 0; st >>= 1) {
    if (o < st) { red0[o] += red0[o + st]; red1[o] += red1[o + st]; }
    __syncthreads();
  }
  if (o == 0) {
    out[(size_t)n * 2 + 0] = red0[0] + fcb[0];
    out[(size_t)n * 2 + 1] = red1[0] + fcb[1];
  }
}

// ---------------------------------------------------------------------------
extern "C" void kernel_launch(void* const* d_in, const int* in_sizes, int n_in,
                              void* d_out, int out_size, void* d_ws, size_t ws_size,
                              hipStream_t stream) {
  const int*   x      = (const int*)  d_in[0];
  const float* embed  = (const float*)d_in[1];
  const float* coefs1 = (const float*)d_in[2];
  const float* bias1  = (const float*)d_in[3];
  const float* wout1  = (const float*)d_in[4];
  const float* bout1  = (const float*)d_in[5];
  const float* coefs2 = (const float*)d_in[6];
  const float* bias2  = (const float*)d_in[7];
  const float* wout2  = (const float*)d_in[8];
  const float* bout2  = (const float*)d_in[9];
  const float* fcw    = (const float*)d_in[10];
  const float* fcb    = (const float*)d_in[11];

  _Float16* bpack = (_Float16*)d_ws;
  float* Hbuf = (float*)((char*)d_ws + (size_t)2 * BPACK_PER_NET * sizeof(_Float16));

  // 1) repack coefs into WMMA-B fragment order (2 * 65536 halfs)
  pack_coefs_kernel<<<(2 * BPACK_PER_NET) / 256, 256, 0, stream>>>(
      coefs1, coefs2, bpack);

  // 2) batched 128x512x128 WMMA GEMM + fused relu/masked-mean per (n, net)
  eq3_main_kernel<<<dim3(BATCH, 2), 256, 0, stream>>>(
      x, embed, bpack, bias1, bias2, Hbuf);

  // 3) tiny epilogue
  head_kernel<<<BATCH, 128, 0, stream>>>(
      Hbuf, wout1, bout1, wout2, bout2, fcw, fcb, (float*)d_out);
}
